// MessageGNNDecoder_89361089560933
// MI455X (gfx1250) — compile-verified, matrix operand order checked by reading, not run on hardware
//
#include <hip/hip_runtime.h>
#include <stdint.h>

// ---------------- constants ----------------
#define BQ 4
#define VQ 2048
#define MQ 8192
#define HQ 64
#define LQ 5
#define TQ 4
#define BH 256   // B*H

// ---------------- types / helpers ----------------
typedef __bf16 bf16_t;
typedef bf16_t v16bf __attribute__((ext_vector_type(16)));
typedef float  v8f   __attribute__((ext_vector_type(8)));

union ABu {
    unsigned int u[8];
    uint4        q[2];
    v16bf        v;
};

__device__ __forceinline__ unsigned int f2bfbits(float x) {
    unsigned int u = __float_as_uint(x);
    return (u + 0x7fffu + ((u >> 16) & 1u)) >> 16;   // round-to-nearest-even
}
__device__ __forceinline__ unsigned short f2bf(float x) { return (unsigned short)f2bfbits(x); }
__device__ __forceinline__ unsigned int pk2bf(float a, float b) {
    return f2bfbits(a) | (f2bfbits(b) << 16);
}

// A-fragment loaders: bf16 direct, or fp32 with on-the-fly conversion (fallback)
__device__ __forceinline__ void loadAfrag(ABu& af, const unsigned short* ar) {
    af.q[0] = *(const uint4*)(ar);
    af.q[1] = *(const uint4*)(ar + 16);
}
__device__ __forceinline__ void loadAfrag(ABu& af, const float* ar) {
    float4 a0 = *(const float4*)(ar);
    float4 a1 = *(const float4*)(ar + 4);
    float4 a2 = *(const float4*)(ar + 16);
    float4 a3 = *(const float4*)(ar + 20);
    af.u[0] = pk2bf(a0.x, a0.y); af.u[1] = pk2bf(a0.z, a0.w);
    af.u[2] = pk2bf(a1.x, a1.y); af.u[3] = pk2bf(a1.z, a1.w);
    af.u[4] = pk2bf(a2.x, a2.y); af.u[5] = pk2bf(a2.z, a2.w);
    af.u[6] = pk2bf(a3.x, a3.y); af.u[7] = pk2bf(a3.z, a3.w);
}

// ---------------- one-shot adjacency fp32 -> bf16 conversion ----------------
// dst[0 .. M^2)  = Av, dst[M^2 .. 2M^2) = Ac
__global__ void aconv(const float* __restrict__ Av, const float* __restrict__ Ac,
                      unsigned short* __restrict__ dst) {
    size_t idx = ((size_t)blockIdx.x * 256 + threadIdx.x) * 8;
    const size_t half = (size_t)MQ * MQ;
    const float* src;
    size_t o;
    if (idx < half) { src = Av; o = idx; } else { src = Ac; o = idx - half; }
    float4 a0 = *(const float4*)(src + o);
    float4 a1 = *(const float4*)(src + o + 4);
    uint4 w;
    w.x = pk2bf(a0.x, a0.y);
    w.y = pk2bf(a0.z, a0.w);
    w.z = pk2bf(a1.x, a1.y);
    w.w = pk2bf(a1.z, a1.w);
    *(uint4*)(dst + idx) = w;
}

// ---------------- weight prep: transpose + fp32->bf16 ----------------
__global__ void wprep(const float* __restrict__ w1v, const float* __restrict__ w1c,
                      const float* __restrict__ w2v, const float* __restrict__ w2c,
                      unsigned short* __restrict__ w1vT, unsigned short* __restrict__ w1cT,
                      unsigned short* __restrict__ w2vT, unsigned short* __restrict__ w2cT) {
    int idx = blockIdx.x * 256 + threadIdx.x;        // 0 .. 122879
    int l = idx / 24576, r = idx % 24576;
    if (r < 16384) {
        const float* src = (r < 8192) ? w1v : w1c;
        unsigned short* dst = (r < 8192) ? w1vT : w1cT;
        int rr = r & 8191;
        int n = rr >> 7, k = rr & 127;
        dst[(size_t)(l * 64 + n) * 128 + k] = f2bf(src[(size_t)(l * 128 + k) * 64 + n]);
    } else {
        int rr = r - 16384;
        const float* src = (rr < 4096) ? w2v : w2c;
        unsigned short* dst = (rr < 4096) ? w2vT : w2cT;
        rr &= 4095;
        int n = rr >> 6, k = rr & 63;
        dst[(size_t)(l * 64 + n) * 64 + k] = f2bf(src[(size_t)(l * 64 + k) * 64 + n]);
    }
}

// ---------------- feat init ----------------
__global__ void feat_init(const float* __restrict__ llr, const int* __restrict__ mapv,
                          const float* __restrict__ w_in, const float* __restrict__ b_in,
                          float* __restrict__ feat) {
    int idx = blockIdx.x * 256 + threadIdx.x;        // 0 .. M*256-1
    int m = idx >> 8, c = idx & 255;
    int b = c >> 6, h = c & 63;
    float x = llr[b * VQ + mapv[m]];
    feat[idx] = x * w_in[h] + b_in[h];
}

__global__ void zero_acc(float* __restrict__ acc) {
    acc[blockIdx.x * 256 + threadIdx.x] = 0.0f;
}

// ---------------- comb: comb_bf16[m][c] and combT_bf16[c][m] ----------------
__global__ __launch_bounds__(256) void comb_kernel(const float* __restrict__ feat,
                                                   const int* __restrict__ types,
                                                   const float* __restrict__ te,  // layer slice [T][H]
                                                   unsigned short* __restrict__ comb,
                                                   unsigned short* __restrict__ combT) {
    __shared__ unsigned short tile[64][264];
    const int t = threadIdx.x;
    const int mbase = blockIdx.x * 64;
    for (int ml = 0; ml < 64; ++ml) {
        int m = mbase + ml;
        int ty = types[m];
        ty = ty < 0 ? 0 : (ty > TQ - 1 ? TQ - 1 : ty);
        float v = feat[(size_t)m * BH + t] + te[ty * HQ + (t & 63)];
        unsigned short us = f2bf(v);
        comb[(size_t)m * BH + t] = us;
        tile[ml][t] = us;
    }
    __syncthreads();
    const int cq = t >> 6;
    const int ml = t & 63;
    for (int cc = 0; cc < 256; cc += 4) {
        int c = cc + cq;
        combT[(size_t)c * MQ + mbase + ml] = tile[ml][c];
    }
}

// ---------------- adjacency GEMM: Y[m][c] = sum_k A[m][k] * X[k][c] ----------------
// grid (64, 2): y selects (Av->Yv) or (Ac->Yc). Block = 8 waves, MBLK=128, N=256, KBLK=64.
template <typename AT>
__global__ __launch_bounds__(256) void adj_gemm(const AT* __restrict__ Amv,
                                                const AT* __restrict__ Amc,
                                                const unsigned short* __restrict__ XT,  // [256][8192] bf16
                                                unsigned short* __restrict__ Yv,
                                                unsigned short* __restrict__ Yc) {
    __shared__ __align__(16) unsigned char xtile[2][32768];  // 2 x (256 rows x 64 bf16)

    const AT* A = (blockIdx.y == 0) ? Amv : Amc;
    unsigned short* Y = (blockIdx.y == 0) ? Yv : Yc;

    const int tid = threadIdx.x;
    const int wave = tid >> 5, lane = tid & 31;
    const int mbase = blockIdx.x * 128 + wave * 16;
    const int mrow = mbase + (lane & 15);
    const int koff0 = (lane & 16) ? 8 : 0;   // documented 16-bit A/B lane K-run split
    const int rsel  = (lane & 16) ? 8 : 0;

    // async producer: thread tid owns XT row tid; 128B column-slice per K-tile
    const unsigned char* gsrc = (const unsigned char*)XT + (size_t)tid * (size_t)(MQ * 2);
    const unsigned int lds0 = (unsigned int)(uintptr_t)(&xtile[0][0]) + (unsigned int)tid * 128u;

    v8f acc[16] = {};

    auto issue = [&](int kb, int buf) {
        const unsigned char* g = gsrc + (size_t)kb * 128u;
        unsigned int l = lds0 + (unsigned int)buf * 32768u;
        // instruction offset applies to BOTH the global and the LDS address (ISA 08 4.4)
        asm volatile(
            "global_load_async_to_lds_b128 %0, %1, off\n\t"
            "global_load_async_to_lds_b128 %0, %1, off offset:16\n\t"
            "global_load_async_to_lds_b128 %0, %1, off offset:32\n\t"
            "global_load_async_to_lds_b128 %0, %1, off offset:48\n\t"
            "global_load_async_to_lds_b128 %0, %1, off offset:64\n\t"
            "global_load_async_to_lds_b128 %0, %1, off offset:80\n\t"
            "global_load_async_to_lds_b128 %0, %1, off offset:96\n\t"
            "global_load_async_to_lds_b128 %0, %1, off offset:112"
            :: "v"(l), "v"(g) : "memory");
    };

    issue(0, 0);
#pragma unroll 2
    for (int kb = 0; kb < MQ / 64; ++kb) {
        const int cb = kb & 1;
        if (kb + 1 < MQ / 64) {
            issue(kb + 1, cb ^ 1);
            asm volatile("s_wait_asynccnt 0x8" ::: "memory");  // current tile done (in-order)
        } else {
            asm volatile("s_wait_asynccnt 0x0" ::: "memory");
        }
        __syncthreads();

        const unsigned char* xb = &xtile[cb][0];
#pragma unroll
        for (int ks = 0; ks < 2; ++ks) {
            ABu af;
            loadAfrag(af, A + (size_t)mrow * MQ + (size_t)kb * 64 + ks * 32 + koff0);

            const unsigned char* bbase = xb + (lane & 15) * 128 + ks * 64 + koff0 * 2;
            ABu bf[2];
            bf[0].q[0] = *(const uint4*)(bbase);
            bf[0].q[1] = *(const uint4*)(bbase + 32);
#pragma unroll
            for (int j = 0; j < 16; ++j) {
                if (j + 1 < 16) {   // double-buffer B fragments: overlap ds_load with wmma
                    const unsigned char* brow = bbase + (j + 1) * 2048;
                    bf[(j + 1) & 1].q[0] = *(const uint4*)(brow);
                    bf[(j + 1) & 1].q[1] = *(const uint4*)(brow + 32);
                }
                acc[j] = __builtin_amdgcn_wmma_f32_16x16x32_bf16(false, af.v, false, bf[j & 1].v,
                                                                 (short)0, acc[j], false, false);
            }
        }
        __syncthreads();
    }

#pragma unroll
    for (int j = 0; j < 16; ++j) {
        const int col = j * 16 + (lane & 15);
#pragma unroll
        for (int v = 0; v < 8; ++v) {
            const int row = mbase + v + rsel;
            Y[(size_t)row * BH + col] = f2bf(acc[j][v]);
        }
    }
}

// ---------------- MLP branch: out += relu([comb|x] @ w1 + b1) @ w2 + b2 ----------------
__device__ __forceinline__ void mlp_branch(const unsigned short* __restrict__ combp,
                                           const unsigned short* __restrict__ xp,
                                           const unsigned short* __restrict__ w1T,
                                           const float* __restrict__ b1,
                                           const unsigned short* __restrict__ w2T,
                                           const float* __restrict__ b2,
                                           int rowbase, int lane,
                                           unsigned short* hrow, v8f out[4]) {
    const int mr = lane & 15;
    const int koff0 = (lane & 16) ? 8 : 0;
    const int rsel  = (lane & 16) ? 8 : 0;

    v8f h[4] = {};
#pragma unroll
    for (int kt = 0; kt < 4; ++kt) {            // K = 128: kt 0,1 from comb; 2,3 from x
        const unsigned short* src = (kt < 2) ? combp : xp;
        const int kcol = (kt & 1) * 32;
        const unsigned short* arow = src + (size_t)(rowbase + mr) * HQ + kcol + koff0;
        ABu a;
        a.q[0] = *(const uint4*)(arow);
        a.q[1] = *(const uint4*)(arow + 16);
#pragma unroll
        for (int j = 0; j < 4; ++j) {
            const unsigned short* brow = w1T + (size_t)(j * 16 + mr) * 128 + kt * 32 + koff0;
            ABu bb;
            bb.q[0] = *(const uint4*)(brow);
            bb.q[1] = *(const uint4*)(brow + 16);
            h[j] = __builtin_amdgcn_wmma_f32_16x16x32_bf16(false, a.v, false, bb.v,
                                                           (short)0, h[j], false, false);
        }
    }
    // bias + relu, transpose via wave-private LDS to build stage-2 A fragments
#pragma unroll
    for (int j = 0; j < 4; ++j) {
        float bias = b1[j * 16 + mr];
#pragma unroll
        for (int v = 0; v < 8; ++v) {
            float x = h[j][v] + bias;
            x = fmaxf(x, 0.0f);
            hrow[(v + rsel) * HQ + j * 16 + mr] = f2bf(x);
        }
    }
#pragma unroll
    for (int kt = 0; kt < 2; ++kt) {            // K = 64
        const unsigned short* arow = hrow + mr * HQ + kt * 32 + koff0;
        ABu a;
        a.q[0] = *(const uint4*)(arow);
        a.q[1] = *(const uint4*)(arow + 16);
#pragma unroll
        for (int j = 0; j < 4; ++j) {
            const unsigned short* brow = w2T + (size_t)(j * 16 + mr) * 64 + kt * 32 + koff0;
            ABu bb;
            bb.q[0] = *(const uint4*)(brow);
            bb.q[1] = *(const uint4*)(brow + 16);
            out[j] = __builtin_amdgcn_wmma_f32_16x16x32_bf16(false, a.v, false, bb.v,
                                                             (short)0, out[j], false, false);
        }
    }
#pragma unroll
    for (int j = 0; j < 4; ++j) {
        float bias = b2[j * 16 + mr];
#pragma unroll
        for (int v = 0; v < 8; ++v) out[j][v] += bias;
    }
}

__global__ __launch_bounds__(256) void mlp_kernel(const unsigned short* __restrict__ comb,
                                                  const unsigned short* __restrict__ v2c,
                                                  const unsigned short* __restrict__ c2v,
                                                  const unsigned short* __restrict__ w1vT, const float* __restrict__ b1v,
                                                  const unsigned short* __restrict__ w2vT, const float* __restrict__ b2v,
                                                  const unsigned short* __restrict__ w1cT, const float* __restrict__ b1c,
                                                  const unsigned short* __restrict__ w2cT, const float* __restrict__ b2c,
                                                  float* __restrict__ feat, int add_res) {
    __shared__ __align__(16) unsigned short hbuf[8][16 * 64];  // wave-private 16x64 tiles
    const int wave = threadIdx.x >> 5, lane = threadIdx.x & 31;
    const int rowbase = (blockIdx.x * 8 + wave) * 16;
    unsigned short* hrow = &hbuf[wave][0];
    const int rsel = (lane & 16) ? 8 : 0;

    v8f uv[4] = {}, uc[4] = {};
    mlp_branch(comb, v2c, w1vT, b1v, w2vT, b2v, rowbase, lane, hrow, uv);
    mlp_branch(comb, c2v, w1cT, b1c, w2cT, b2c, rowbase, lane, hrow, uc);

#pragma unroll
    for (int j = 0; j < 4; ++j) {
        const int col = j * 16 + (lane & 15);
#pragma unroll
        for (int v = 0; v < 8; ++v) {
            const size_t idx = (size_t)(rowbase + v + rsel) * HQ + col;
            float x = uv[j][v] + uc[j][v];
            if (add_res) x += feat[idx];
            feat[idx] = x;
        }
    }
}

// ---------------- decode + segment_sum (atomic) ----------------
__global__ __launch_bounds__(256) void decode_reduce(const float* __restrict__ feat,
                                                     const int* __restrict__ mapv,
                                                     const float* __restrict__ proj_w,
                                                     const float* __restrict__ proj_b,
                                                     float* __restrict__ acc) {
    const int wave = threadIdx.x >> 5, lane = threadIdx.x & 31;
    const int gw = blockIdx.x * 8 + wave;
    const float pw0 = proj_w[lane], pw1 = proj_w[lane + 32];
    const float pb = proj_b[0];
    for (int rr = 0; rr < 32; ++rr) {
        const int r = gw * 32 + rr;                 // r = m*B + b
        const float* fr = feat + (size_t)r * HQ;
        float p = fr[lane] * pw0 + fr[lane + 32] * pw1;
#pragma unroll
        for (int off = 16; off > 0; off >>= 1) p += __shfl_xor(p, off, 32);
        if (lane == 0) {
            const int m = r >> 2, b = r & 3;
            atomicAdd(&acc[b * VQ + mapv[m]], p + pb);
        }
    }
}

__global__ void final_sigmoid(const float* __restrict__ acc, const float* __restrict__ llr,
                              float* __restrict__ out) {
    int i = blockIdx.x * 256 + threadIdx.x;
    float x = acc[i] + llr[i];
    out[i] = 1.0f / (1.0f + __expf(-x));
}

// ---------------- launcher ----------------
extern "C" void kernel_launch(void* const* d_in, const int* in_sizes, int n_in,
                              void* d_out, int out_size, void* d_ws, size_t ws_size,
                              hipStream_t stream) {
    (void)in_sizes; (void)n_in; (void)out_size;

    const float* llr   = (const float*)d_in[0];
    const int*   mapv  = (const int*)d_in[1];
    const int*   types = (const int*)d_in[2];
    const float* Av    = (const float*)d_in[3];   // var_to_check
    const float* Ac    = (const float*)d_in[4];   // check_to_var
    const float* w_in  = (const float*)d_in[5];
    const float* b_in  = (const float*)d_in[6];
    const float* te    = (const float*)d_in[7];
    const float* w1v   = (const float*)d_in[8];
    const float* b1v   = (const float*)d_in[9];
    const float* w2v   = (const float*)d_in[10];
    const float* b2v   = (const float*)d_in[11];
    const float* w1c   = (const float*)d_in[12];
    const float* b1c   = (const float*)d_in[13];
    const float* w2c   = (const float*)d_in[14];
    const float* b2c   = (const float*)d_in[15];
    const float* pw    = (const float*)d_in[16];
    const float* pb    = (const float*)d_in[17];
    float* out = (float*)d_out;

    char* ws = (char*)d_ws;
    size_t off = 0;
    auto wsalloc = [&](size_t bytes) -> char* {
        char* p = ws + off;
        off += (bytes + 255) & ~(size_t)255;
        return p;
    };
    float*          feat  = (float*)wsalloc((size_t)MQ * BH * 4);
    unsigned short* comb  = (unsigned short*)wsalloc((size_t)MQ * BH * 2);
    unsigned short* combT = (unsigned short*)wsalloc((size_t)BH * MQ * 2);
    unsigned short* v2c   = (unsigned short*)wsalloc((size_t)MQ * BH * 2);
    unsigned short* c2v   = (unsigned short*)wsalloc((size_t)MQ * BH * 2);
    float*          acc   = (float*)wsalloc((size_t)BQ * VQ * 4);
    unsigned short* w1vT  = (unsigned short*)wsalloc((size_t)LQ * 64 * 128 * 2);
    unsigned short* w1cT  = (unsigned short*)wsalloc((size_t)LQ * 64 * 128 * 2);
    unsigned short* w2vT  = (unsigned short*)wsalloc((size_t)LQ * 64 * 64 * 2);
    unsigned short* w2cT  = (unsigned short*)wsalloc((size_t)LQ * 64 * 64 * 2);

    // optional big buffer: adjacency matrices as bf16 (268 MB) -> halves per-layer A traffic
    const size_t abf_bytes = (size_t)2 * MQ * MQ * 2;
    const bool use_abf = (off + abf_bytes) <= ws_size;
    unsigned short* Abf = use_abf ? (unsigned short*)wsalloc(abf_bytes) : nullptr;

    wprep<<<480, 256, 0, stream>>>(w1v, w1c, w2v, w2c, w1vT, w1cT, w2vT, w2cT);
    feat_init<<<MQ * BH / 256, 256, 0, stream>>>(llr, mapv, w_in, b_in, feat);
    zero_acc<<<BQ * VQ / 256, 256, 0, stream>>>(acc);
    if (use_abf) {
        aconv<<<(2 * (size_t)MQ * MQ) / (8 * 256), 256, 0, stream>>>(Av, Ac, Abf);
    }

    for (int l = 0; l < LQ; ++l) {
        comb_kernel<<<MQ / 64, 256, 0, stream>>>(feat, types, te + (size_t)l * TQ * HQ, comb, combT);
        if (use_abf) {
            adj_gemm<unsigned short><<<dim3(MQ / 128, 2), 256, 0, stream>>>(
                Abf, Abf + (size_t)MQ * MQ, combT, v2c, c2v);
        } else {
            adj_gemm<float><<<dim3(MQ / 128, 2), 256, 0, stream>>>(Av, Ac, combT, v2c, c2v);
        }
        mlp_kernel<<<(MQ * BQ) / 128, 256, 0, stream>>>(
            comb, v2c, c2v,
            w1vT + (size_t)l * 64 * 128, b1v + l * HQ,
            w2vT + (size_t)l * 64 * 64,  b2v + l * HQ,
            w1cT + (size_t)l * 64 * 128, b1c + l * HQ,
            w2cT + (size_t)l * 64 * 64,  b2c + l * HQ,
            feat, l > 0);
    }

    decode_reduce<<<(MQ * BQ) / (32 * 8), 256, 0, stream>>>(feat, mapv, pw, pb, acc);
    final_sigmoid<<<BQ * VQ / 256, 256, 0, stream>>>(acc, llr, out);
}
